// RNN_29102698398242
// MI455X (gfx1250) — matrix operand order repaired and checked by reading, not compiled, not run on hardware
//
#include <hip/hip_runtime.h>
#include <stdint.h>

// Problem sizes (match reference)
static constexpr int kH = 1024;
static constexpr int kV = 4096;
static constexpr int kB = 64;
static constexpr int kT = 512;

typedef __attribute__((ext_vector_type(16))) __bf16 v16bf;
typedef __attribute__((ext_vector_type(8)))  float  v8f;

union Frag16 {
    v16bf v;
    int4  q[2];
};

__device__ __forceinline__ unsigned short f32_to_bf16(float f) {
    union { float f; unsigned int u; } x;
    x.f = f;
    unsigned int u = x.u;
    u += 0x7FFFu + ((u >> 16) & 1u);   // round-to-nearest-even
    return (unsigned short)(u >> 16);
}

// ---------------------------------------------------------------------------
// f32 -> bf16 conversion (weights)
// ---------------------------------------------------------------------------
__global__ void cvt_bf16_kernel(const float* __restrict__ src,
                                unsigned short* __restrict__ dst, int n) {
    int i = blockIdx.x * blockDim.x + threadIdx.x;
    if (i < n) dst[i] = f32_to_bf16(src[i]);
}

// ---------------------------------------------------------------------------
// One recurrence step: h_t = tanh(x_t + h_{t-1} @ Whh^T + b)
// Grid: 32 blocks x 256 threads = 256 waves; wave -> one 16x16 tile of (B=64, H=1024).
// ---------------------------------------------------------------------------
__global__ __launch_bounds__(256) void rnn_step_kernel(
    unsigned short* __restrict__ Hs,          // (B,T,H) bf16: read t-1, write t
    const unsigned short* __restrict__ Whh,   // (H,H) bf16
    const int* __restrict__ X,                // (B,T)
    const float* __restrict__ Wxh,            // (V,H) f32
    const float* __restrict__ bias,           // (H)
    float* __restrict__ hid_out,              // (H,T,B) f32
    int t)
{
    const int lane  = threadIdx.x & 31;
    const int wave  = threadIdx.x >> 5;
    const int gw    = blockIdx.x * 8 + wave;   // 0..255
    const int mt    = gw >> 6;                 // batch tile 0..3
    const int nt    = gw & 63;                 // hidden tile 0..63
    const int mlo   = lane & 15;
    const int khalf = lane >> 4;

    v8f acc = {0.f, 0.f, 0.f, 0.f, 0.f, 0.f, 0.f, 0.f};

    if (t > 0) {
        const int b_row = mt * 16 + mlo;
        const unsigned short* aRow = Hs + ((size_t)b_row * kT + (t - 1)) * kH;
        const unsigned short* bRow = Whh + (size_t)(nt * 16 + mlo) * kH + khalf * 8;
        #pragma unroll 4
        for (int k = 0; k < kH; k += 32) {
            Frag16 a, bm;
            // A 16x32 bf16 layout: lanes 0-15 K=0..7,16..23 ; lanes 16-31 K=8..15,24..31
            a.q[0] = *(const int4*)(aRow + k + khalf * 8);
            a.q[1] = *(const int4*)(aRow + k + khalf * 8 + 16);
            // B 32x16 bf16 layout: lane n=lane&15, lanes 0-15 K=0..7,16..23 ; lanes 16-31 K=8..15,24..31
            bm.q[0] = *(const int4*)(bRow + k);
            bm.q[1] = *(const int4*)(bRow + k + 16);
            acc = __builtin_amdgcn_wmma_f32_16x16x32_bf16(
                false, a.v, false, bm.v, (short)0, acc, false, false);
        }
    }

    // C/D layout: VGPR r -> M = r + 8*(lane>>4), N = lane&15
    const int j  = nt * 16 + mlo;
    const float bj = bias[j];
    #pragma unroll
    for (int r = 0; r < 8; ++r) {
        const int b_idx = mt * 16 + r + khalf * 8;
        const int xi = X[b_idx * kT + t];
        const float val = tanhf(acc[r] + Wxh[(size_t)xi * kH + j] + bj);
        Hs[((size_t)b_idx * kT + t) * kH + j] = f32_to_bf16(val);
        hid_out[(size_t)j * (kT * kB) + (size_t)t * kB + b_idx] = val;  // (H,T,B)
    }
}

// ---------------------------------------------------------------------------
// Logits GEMM: Z(B*T, V) = Hs(B*T, H) @ Wvh(V, H)^T + c
// 128x128x32 tiles; 8 waves/WG; each wave 64x32 region = 8 tiles, 8 WMMA/K-step.
// Tile movement via Tensor Data Mover (double-buffered LDS) when available.
// ---------------------------------------------------------------------------
static constexpr int BM = 128, BN = 128, BK = 32;
static constexpr int NK = kH / BK;

#if __has_builtin(__builtin_amdgcn_tensor_load_to_lds)
#define USE_TDM 1
typedef unsigned int u32x4 __attribute__((ext_vector_type(4)));
typedef int          i32x8 __attribute__((ext_vector_type(8)));
typedef int          i32x4 __attribute__((ext_vector_type(4)));

__device__ __forceinline__ unsigned lds_addr_of(const void* p) {
    return (unsigned)(uintptr_t)(__attribute__((address_space(3))) const void*)p;
}

// 2D TDM tile load: tile_k x tile_rows elements of 2 bytes, row stride = stride
// elements. Builds D# groups 0/1 per CDNA5 ISA §8.3-8.4 (groups 2/3/aux zero: 2D).
__device__ __forceinline__ void tdm_load_tile_2d(
    const void* gptr, unsigned lds_addr,
    unsigned tile_k, unsigned tile_rows,
    unsigned stride_elems, unsigned tensor_d0, unsigned tensor_d1)
{
    const unsigned long long ga = (unsigned long long)(uintptr_t)gptr;
    u32x4 g0;
    g0[0] = 1u;                                        // count=1, is_restore=0
    g0[1] = lds_addr;                                  // [63:32] lds byte addr
    g0[2] = (unsigned)(ga & 0xFFFFFFFFu);              // [95:64] global addr lo
    g0[3] = (unsigned)((ga >> 32) & 0x01FFFFFFu)       // [120:96] global addr hi
            | (2u << 30);                              // [127:126] type=2 (image)
    i32x8 g1;
    g1[0] = (int)(1u << 16);                           // data_size=1 (2 bytes)
    g1[1] = (int)((tensor_d0 & 0xFFFFu) << 16);        // [63:48] tensor_dim0 lo
    g1[2] = (int)((tensor_d0 >> 16) |                  // [79:64] tensor_dim0 hi
                  ((tensor_d1 & 0xFFFFu) << 16));      // [95:80] tensor_dim1 lo
    g1[3] = (int)((tensor_d1 >> 16) |                  // [111:96] tensor_dim1 hi
                  (tile_k << 16));                     // [127:112] tile_dim0
    g1[4] = (int)tile_rows;                            // [143:128] tile_dim1
    g1[5] = (int)stride_elems;                         // [191:160] dim0 stride lo
    g1[6] = 0;
    g1[7] = 0;
    const i32x4 z4 = {0, 0, 0, 0};
    const i32x8 z8 = {0, 0, 0, 0, 0, 0, 0, 0};
    // 6-arg toolchain variant: (g0, g1, g2, g3, aux, cpol)
    __builtin_amdgcn_tensor_load_to_lds(g0, g1, z4, z4, z8, 0);
}
#else
#define USE_TDM 0
#endif

__global__ __launch_bounds__(256) void logits_gemm_kernel(
    const unsigned short* __restrict__ A,   // (B*T, H) bf16
    const unsigned short* __restrict__ Bw,  // (V, H)   bf16
    const float* __restrict__ cbias,        // (V)
    float* __restrict__ Z)                  // (B*T, V) f32
{
#if USE_TDM
    __shared__ __align__(16) unsigned short As[2][BM * BK];
    __shared__ __align__(16) unsigned short Bs[2][BN * BK];
#else
    __shared__ __align__(16) unsigned short As[1][BM * BK];
    __shared__ __align__(16) unsigned short Bs[1][BN * BK];
#endif

    const int tid   = threadIdx.x;
    const int lane  = tid & 31;
    const int wave  = tid >> 5;
    const int mwave = wave >> 2;   // 0..1 -> 64-row half
    const int nwave = wave & 3;    // 0..3 -> 32-col slice
    const int mlo   = lane & 15;
    const int khalf = lane >> 4;

    const int m0 = blockIdx.y * BM;
    const int n0 = blockIdx.x * BN;

    v8f acc[8];
    #pragma unroll
    for (int i = 0; i < 8; ++i) {
        v8f zz = {0.f, 0.f, 0.f, 0.f, 0.f, 0.f, 0.f, 0.f};
        acc[i] = zz;
    }

#if USE_TDM
    // ---- TDM double-buffered pipeline: DMA tile k+1 while computing tile k ----
    if (wave == 0)
        tdm_load_tile_2d(A + (size_t)m0 * kH, lds_addr_of(&As[0][0]),
                         BK, BM, kH, kH, (unsigned)(kB * kT));
    if (wave == 1)
        tdm_load_tile_2d(Bw + (size_t)n0 * kH, lds_addr_of(&Bs[0][0]),
                         BK, BN, kH, kH, (unsigned)kV);

    for (int kt = 0; kt < NK; ++kt) {
        const int cur = kt & 1;
        if (kt + 1 < NK) {
            const int k0n = (kt + 1) * BK;
            if (wave == 0)
                tdm_load_tile_2d(A + (size_t)m0 * kH + k0n,
                                 lds_addr_of(&As[cur ^ 1][0]),
                                 BK, BM, kH, kH, (unsigned)(kB * kT));
            if (wave == 1)
                tdm_load_tile_2d(Bw + (size_t)n0 * kH + k0n,
                                 lds_addr_of(&Bs[cur ^ 1][0]),
                                 BK, BN, kH, kH, (unsigned)kV);
            __builtin_amdgcn_s_wait_tensorcnt(1);   // stage kt landed, kt+1 in flight
        } else {
            __builtin_amdgcn_s_wait_tensorcnt(0);   // final stage landed
        }
        __syncthreads();
#else
    // ---- Fallback: synchronous per-thread staging ----
    for (int kt = 0; kt < NK; ++kt) {
        const int cur = 0;
        const int k0 = kt * BK;
        const int srow = tid >> 1;          // 128 rows, 2 threads/row
        const int scol = (tid & 1) * 16;    // 16 bf16 (32B) per thread
        __syncthreads();
        {
            const unsigned short* sa = A  + (size_t)(m0 + srow) * kH + k0 + scol;
            const unsigned short* sb = Bw + (size_t)(n0 + srow) * kH + k0 + scol;
            int4 a0 = ((const int4*)sa)[0];
            int4 a1 = ((const int4*)sa)[1];
            int4 b0 = ((const int4*)sb)[0];
            int4 b1 = ((const int4*)sb)[1];
            ((int4*)&As[0][srow * BK + scol])[0] = a0;
            ((int4*)&As[0][srow * BK + scol])[1] = a1;
            ((int4*)&Bs[0][srow * BK + scol])[0] = b0;
            ((int4*)&Bs[0][srow * BK + scol])[1] = b1;
        }
        __syncthreads();
#endif

        Frag16 af[4];
        #pragma unroll
        for (int m = 0; m < 4; ++m) {
            const unsigned short* ap =
                &As[cur][(mwave * 64 + m * 16 + mlo) * BK + khalf * 8];
            af[m].q[0] = *(const int4*)ap;
            af[m].q[1] = *(const int4*)(ap + 16);
        }
        Frag16 bf2[2];
        #pragma unroll
        for (int n = 0; n < 2; ++n) {
            const unsigned short* bp =
                &Bs[cur][(nwave * 32 + n * 16 + mlo) * BK + khalf * 8];
            bf2[n].q[0] = *(const int4*)bp;
            bf2[n].q[1] = *(const int4*)(bp + 16);
        }
        #pragma unroll
        for (int m = 0; m < 4; ++m)
            #pragma unroll
            for (int n = 0; n < 2; ++n)
                acc[m * 2 + n] = __builtin_amdgcn_wmma_f32_16x16x32_bf16(
                    false, af[m].v, false, bf2[n].v, (short)0,
                    acc[m * 2 + n], false, false);

#if USE_TDM
        __syncthreads();   // all waves done reading buf[cur] before it is re-filled
    }
#else
    }
#endif

    #pragma unroll
    for (int m = 0; m < 4; ++m) {
        #pragma unroll
        for (int n = 0; n < 2; ++n) {
            const v8f a = acc[m * 2 + n];
            const int col = n0 + nwave * 32 + n * 16 + mlo;
            const float cb = cbias[col];
            #pragma unroll
            for (int r = 0; r < 8; ++r) {
                const int row = m0 + mwave * 64 + m * 16 + r + khalf * 8;
                Z[(size_t)row * kV + col] = a[r] + cb;
            }
        }
    }
}

// ---------------------------------------------------------------------------
// In-place log-softmax over the last dim (V=4096), one block per (b,t) row.
// ---------------------------------------------------------------------------
__global__ __launch_bounds__(256) void logsoftmax_kernel(float* __restrict__ Z) {
    float* p = Z + (size_t)blockIdx.x * kV;
    const int tid = threadIdx.x;

    float v[16];
    float mx = -__builtin_inff();
    #pragma unroll
    for (int i = 0; i < 16; ++i) {
        v[i] = p[tid + i * 256];
        mx = fmaxf(mx, v[i]);
    }
    #pragma unroll
    for (int off = 16; off > 0; off >>= 1)
        mx = fmaxf(mx, __shfl_xor(mx, off, 32));

    __shared__ float redm[8];
    __shared__ float reds[8];
    if ((tid & 31) == 0) redm[tid >> 5] = mx;
    __syncthreads();
    float bm = redm[0];
    #pragma unroll
    for (int w = 1; w < 8; ++w) bm = fmaxf(bm, redm[w]);

    float s = 0.f;
    #pragma unroll
    for (int i = 0; i < 16; ++i) s += __expf(v[i] - bm);
    #pragma unroll
    for (int off = 16; off > 0; off >>= 1)
        s += __shfl_xor(s, off, 32);
    if ((tid & 31) == 0) reds[tid >> 5] = s;
    __syncthreads();
    float ts = 0.f;
    #pragma unroll
    for (int w = 0; w < 8; ++w) ts += reds[w];

    const float lse = bm + __logf(ts);
    #pragma unroll
    for (int i = 0; i < 16; ++i)
        p[tid + i * 256] = v[i] - lse;
}

// ---------------------------------------------------------------------------
extern "C" void kernel_launch(void* const* d_in, const int* in_sizes, int n_in,
                              void* d_out, int out_size, void* d_ws, size_t ws_size,
                              hipStream_t stream) {
    const int*   X     = (const int*)d_in[0];
    const float* Wxh   = (const float*)d_in[1];
    const float* Whh   = (const float*)d_in[2];
    const float* Wvh   = (const float*)d_in[3];
    const float* bias  = (const float*)d_in[4];
    const float* cbias = (const float*)d_in[5];

    char* ws = (char*)d_ws;
    unsigned short* Whh_bf = (unsigned short*)ws;                              // 2 MB
    unsigned short* Wvh_bf = (unsigned short*)(ws + (size_t)2  * 1024 * 1024); // 8 MB
    unsigned short* Hs_bf  = (unsigned short*)(ws + (size_t)10 * 1024 * 1024); // 64 MB (B,T,H)

    float* outs = (float*)d_out;                       // (B,T,V)
    float* hid  = outs + (size_t)kB * kT * kV;         // (H,T,B)

    cvt_bf16_kernel<<<(kH * kH + 255) / 256, 256, 0, stream>>>(Whh, Whh_bf, kH * kH);
    cvt_bf16_kernel<<<(kV * kH + 255) / 256, 256, 0, stream>>>(Wvh, Wvh_bf, kV * kH);

    for (int t = 0; t < kT; ++t)
        rnn_step_kernel<<<32, 256, 0, stream>>>(Hs_bf, Whh_bf, X, Wxh, bias, hid, t);

    dim3 g(kV / BN, (kB * kT) / BM);
    logits_gemm_kernel<<<g, 256, 0, stream>>>(Hs_bf, Wvh_bf, cbias, outs);

    logsoftmax_kernel<<<kB * kT, 256, 0, stream>>>(outs);
}
